// AudioClassifierBNN_50766513438830
// MI455X (gfx1250) — 8-bit matrix operand order repaired and checked by reading, compiled, not run on hardware
//
#include <hip/hip_runtime.h>
#include <hip/hip_bf16.h>
#include <stdint.h>


typedef __attribute__((ext_vector_type(8))) int v8i;

union V8  { v8i v; int4 q[2]; int i[8]; };
union B16 { int4 v; int8_t b[16]; };

#define NB  128
#define IH  256
#define IW  512
#define OH1 128
#define OW1 256
#define C1  16

#define OH2 65
#define OW2 129
#define M2  (OH2*OW2)
#define MT2 ((M2+15)/16)
#define C2  64

#define OH3 34
#define OW3 66
#define M3  (OH3*OW3)
#define MT3 ((M3+15)/16)
#define C3  32

__device__ __forceinline__ float fsignf(float w){ return (w>0.f)?1.f:((w<0.f)?-1.f:0.f); }
__device__ __forceinline__ int8_t bsign(float w){ return (w>0.f)?(int8_t)1:((w<0.f)?(int8_t)(-1):(int8_t)0); }

__device__ __forceinline__ void async_ld_b128(void* lds_dst, const void* gsrc){
    uint32_t loff = (uint32_t)(uintptr_t)lds_dst;
    asm volatile("global_load_async_to_lds_b128 %0, %1, off"
                 :: "v"(loff), "v"(gsrc) : "memory");
}
__device__ __forceinline__ void wait_async_all(){
    asm volatile("s_wait_asynccnt 0x0" ::: "memory");
}

__global__ void conv1_kernel(const float* __restrict__ x, const float* __restrict__ w1,
                             const float* __restrict__ coef, float* __restrict__ stat,
                             int8_t* __restrict__ a1, int mode)
{
    __shared__ float sw[C1*25];
    __shared__ float ls[C1], lsq[C1];
    int tid = threadIdx.x;
    for (int i = tid; i < C1*25; i += blockDim.x) sw[i] = fsignf(w1[i]);
    if (mode == 0) for (int i = tid; i < C1; i += blockDim.x){ ls[i]=0.f; lsq[i]=0.f; }
    __syncthreads();

    int gid = blockIdx.x*blockDim.x + tid;
    int n   = gid >> 15;
    int pos = gid & 32767;
    int oh  = pos >> 8, ow = pos & 255;

    const float* xi = x + (size_t)n*IH*IW;
    float px[25];
    #pragma unroll
    for (int kh = 0; kh < 5; kh++){
        int ih = oh*2 + kh - 2;
        #pragma unroll
        for (int kw = 0; kw < 5; kw++){
            int iw = ow*2 + kw - 2;
            px[kh*5+kw] = (ih>=0 && ih<IH && iw>=0 && iw<IW) ? xi[(size_t)ih*IW + iw] : 0.f;
        }
    }

    B16 outb;
    #pragma unroll
    for (int c = 0; c < C1; c++){
        float acc = 0.f;
        #pragma unroll
        for (int t = 0; t < 25; t++) acc += sw[c*25+t]*px[t];
        if (mode == 0){
            atomicAdd(&ls[c],  acc);
            atomicAdd(&lsq[c], acc*acc);
        } else {
            float y = acc*coef[c] + coef[C1+c];
            outb.b[c] = (y >= 0.f) ? (int8_t)1 : (int8_t)(-1);
        }
    }
    if (mode == 0){
        __syncthreads();
        for (int i = tid; i < C1; i += blockDim.x){
            atomicAdd(stat+i,    ls[i]);
            atomicAdd(stat+C1+i, lsq[i]);
        }
    } else {
        *(int4*)(a1 + (size_t)gid*16) = outb.v;
    }
}

__global__ void finalize_kernel(const float* __restrict__ stat,
                                const float* __restrict__ gamma,
                                const float* __restrict__ beta,
                                float* __restrict__ coef, float invcnt, int nch)
{
    int i = threadIdx.x;
    if (i < nch){
        float m   = stat[i]*invcnt;
        float var = stat[nch+i]*invcnt - m*m;
        float rs  = rsqrtf(var + 1e-5f);
        float A   = rs*gamma[i];
        coef[i]     = A;
        coef[nch+i] = beta[i] - m*A;
    }
}

__global__ void prep_bfrag(const float* __restrict__ w2, const float* __restrict__ w3,
                           int8_t* __restrict__ bf2, int8_t* __restrict__ bf3)
{
    int tid = threadIdx.x + blockIdx.x*blockDim.x;
    int stride = blockDim.x*gridDim.x;
    for (int i = tid; i < 4*3*32*32; i += stride){
        int byte = i & 3, d = (i>>2)&7, lane = (i>>5)&31, qt = i>>10;
        int q = qt % 3, t = qt / 3;
        int K = q*64 + ((d>>2)*32) + ((lane>=16)?16:0) + (d&3)*4 + byte;
        int oc = t*16 + (lane&15);
        int8_t v = 0;
        if (K < 144){
            int tap = K >> 4, c = K & 15;
            int kh = tap/3, kw = tap%3;
            v = bsign(w2[((oc*C1 + c)*3 + kh)*3 + kw]);
        }
        bf2[i] = v;
    }
    for (int i = tid; i < 2*9*32*32; i += stride){
        int byte = i & 3, d = (i>>2)&7, lane = (i>>5)&31, qt = i>>10;
        int q = qt % 9, t = qt / 9;
        int kc = ((d>>2)*32) + ((lane>=16)?16:0) + (d&3)*4 + byte;
        int oc = t*16 + (lane&15);
        int kh = q/3, kw = q%3;
        bf3[i] = bsign(w3[((oc*C2 + kc)*3 + kh)*3 + kw]);
    }
}

__global__ void __launch_bounds__(32) conv2_kernel(
    const int8_t* __restrict__ a1, const int8_t* __restrict__ bf2,
    const float* __restrict__ coef, float* __restrict__ stat,
    int8_t* __restrict__ a2, int mode)
{
    __shared__ __align__(16) int8_t Araw[2][2][16*64];
    __shared__ float ls[C2], lsq[C2];
    int l  = threadIdx.x;
    int m0 = blockIdx.x*2, n = blockIdx.y;
    const int8_t* img = a1 + (size_t)n*OH1*OW1*C1;

    {
        int opos = m0*16;
        if (opos < M2){
            int oh = opos/OW2; int ih = oh*2 - 2;
            if (ih >= 0) __builtin_prefetch(img + ((size_t)ih*OW1)*C1, 0, 0);
        }
    }

    auto stage = [&](int b, int q){
        #pragma unroll
        for (int k = 0; k < 4; k++){
            int s  = l + k*32;
            int mi = s >> 6, s2 = s & 63;
            int r  = s2 >> 2, p = s2 & 3, tap = q*4 + p;
            int8_t* dst = &Araw[b][mi][s2*16];
            int opos = (m0 + mi)*16 + r;
            bool valid = false;
            const int8_t* src = img;
            if (tap < 9 && opos < M2){
                int oh = opos/OW2, ow = opos%OW2;
                int ih = oh*2 + (tap/3) - 2;
                int iw = ow*2 + (tap%3) - 2;
                if (ih>=0 && ih<OH1 && iw>=0 && iw<OW1){
                    valid = true;
                    src = img + ((size_t)ih*OW1 + iw)*C1;
                }
            }
            if (valid) async_ld_b128(dst, src);
            else       { int4 z = {0,0,0,0}; *(int4*)dst = z; }
        }
    };

    v8i acc0[4] = {}, acc1[4] = {};
    stage(0, 0);
    for (int q = 0; q < 3; q++){
        wait_async_all();
        __syncthreads();
        V8 a0, a1f;
        int r = l & 15, kb = (l>=16)?16:0;
        #pragma unroll
        for (int d = 0; d < 8; d++){
            int koff = ((d>>2)<<5) + ((d&3)<<2) + kb;
            a0.i[d]  = *(const int*)(&Araw[q&1][0][r*64 + koff]);
            a1f.i[d] = *(const int*)(&Araw[q&1][1][r*64 + koff]);
        }
        if (q+1 < 3) stage((q+1)&1, q+1);
        #pragma unroll
        for (int t = 0; t < 4; t++){
            V8 b;
            const int8_t* bp = bf2 + ((size_t)((t*3+q)*32 + l))*32;
            b.q[0] = *(const int4*)(bp);
            b.q[1] = *(const int4*)(bp+16);
            acc0[t] = __builtin_amdgcn_wmma_i32_16x16x64_iu8(true, a0.v,  true, b.v,
                                                             acc0[t], false, false);
            acc1[t] = __builtin_amdgcn_wmma_i32_16x16x64_iu8(true, a1f.v, true, b.v,
                                                             acc1[t], false, false);
        }
    }

    if (mode == 0){
        for (int c = l; c < C2; c += 32){ ls[c]=0.f; lsq[c]=0.f; }
        __syncthreads();
        #pragma unroll
        for (int t = 0; t < 4; t++){
            int c = t*16 + (l&15);
            #pragma unroll
            for (int v = 0; v < 8; v++){
                float f0 = (float)acc0[t][v];
                float f1 = (float)acc1[t][v];
                atomicAdd(&ls[c],  f0 + f1);
                atomicAdd(&lsq[c], f0*f0 + f1*f1);
            }
        }
        __syncthreads();
        for (int c = l; c < C2; c += 32){
            atomicAdd(stat+c,    ls[c]);
            atomicAdd(stat+C2+c, lsq[c]);
        }
    } else {
        #pragma unroll
        for (int t = 0; t < 4; t++){
            int c = t*16 + (l&15);
            float A = coef[c], Bc = coef[C2+c];
            #pragma unroll
            for (int v = 0; v < 8; v++){
                int row = v + ((l>=16)?8:0);
                int p0 = m0*16 + row;
                int p1 = (m0+1)*16 + row;
                if (p0 < M2){
                    float y = (float)acc0[t][v]*A + Bc;
                    a2[((size_t)n*M2 + p0)*C2 + c] = (y>=0.f)?(int8_t)1:(int8_t)(-1);
                }
                if (p1 < M2){
                    float y = (float)acc1[t][v]*A + Bc;
                    a2[((size_t)n*M2 + p1)*C2 + c] = (y>=0.f)?(int8_t)1:(int8_t)(-1);
                }
            }
        }
    }
}

__global__ void __launch_bounds__(32) conv3_kernel(
    const int8_t* __restrict__ a2, const int8_t* __restrict__ bf3,
    const float* __restrict__ coef, float* __restrict__ stat,
    float* __restrict__ pooled, int mode)
{
    __shared__ __align__(16) int8_t Araw[2][16*64];
    __shared__ float ls[C3], lsq[C3];
    __shared__ float lp[C3];
    int l  = threadIdx.x;
    int mt = blockIdx.x, n = blockIdx.y;
    const int8_t* img = a2 + (size_t)n*M2*C2;

    auto stage = [&](int b, int q){
        #pragma unroll
        for (int k = 0; k < 2; k++){
            int s = l + k*32;
            int r = s>>2, p = s&3;
            int8_t* dst = &Araw[b][s*16];
            int opos = mt*16 + r;
            bool valid = false;
            const int8_t* src = img;
            if (opos < M3){
                int oh = opos/OW3, ow = opos%OW3;
                int ih = oh*2 + (q/3) - 2;
                int iw = ow*2 + (q%3) - 2;
                if (ih>=0 && ih<OH2 && iw>=0 && iw<OW2){
                    valid = true;
                    src = img + ((size_t)ih*OW2 + iw)*C2 + p*16;
                }
            }
            if (valid) async_ld_b128(dst, src);
            else       { int4 z = {0,0,0,0}; *(int4*)dst = z; }
        }
    };

    v8i acc[2] = {};
    stage(0, 0);
    for (int q = 0; q < 9; q++){
        wait_async_all();
        __syncthreads();
        V8 a;
        int r = l & 15, kb = (l>=16)?16:0;
        #pragma unroll
        for (int d = 0; d < 8; d++){
            int koff = ((d>>2)<<5) + ((d&3)<<2) + kb;
            a.i[d] = *(const int*)(&Araw[q&1][r*64 + koff]);
        }
        if (q+1 < 9) stage((q+1)&1, q+1);
        #pragma unroll
        for (int t = 0; t < 2; t++){
            V8 b;
            const int8_t* bp = bf3 + ((size_t)((t*9+q)*32 + l))*32;
            b.q[0] = *(const int4*)(bp);
            b.q[1] = *(const int4*)(bp+16);
            acc[t] = __builtin_amdgcn_wmma_i32_16x16x64_iu8(true, a.v, true, b.v,
                                                            acc[t], false, false);
        }
    }

    if (mode == 0){
        for (int c = l; c < C3; c += 32){ ls[c]=0.f; lsq[c]=0.f; }
        __syncthreads();
        #pragma unroll
        for (int t = 0; t < 2; t++){
            int c = t*16 + (l&15);
            #pragma unroll
            for (int v = 0; v < 8; v++){
                float f = (float)acc[t][v];
                atomicAdd(&ls[c],  f);
                atomicAdd(&lsq[c], f*f);
            }
        }
        __syncthreads();
        for (int c = l; c < C3; c += 32){
            atomicAdd(stat+c,    ls[c]);
            atomicAdd(stat+C3+c, lsq[c]);
        }
    } else {
        for (int c = l; c < C3; c += 32) lp[c] = 0.f;
        __syncthreads();
        #pragma unroll
        for (int t = 0; t < 2; t++){
            int c = t*16 + (l&15);
            float A = coef[c], Bc = coef[C3+c];
            #pragma unroll
            for (int v = 0; v < 8; v++){
                int opos = mt*16 + v + ((l>=16)?8:0);
                if (opos < M3){
                    float y = (float)acc[t][v]*A + Bc;
                    atomicAdd(&lp[c], y);
                }
            }
        }
        __syncthreads();
        for (int c = l; c < C3; c += 32)
            atomicAdd(pooled + (size_t)n*C3 + c, lp[c]*(1.0f/(float)M3));
    }
}

__global__ void __launch_bounds__(128) head_kernel(
    const float* __restrict__ pooled, const float* __restrict__ wfc,
    const float* __restrict__ bfc,    const float* __restrict__ g4,
    const float* __restrict__ b4,     float* __restrict__ out)
{
    __shared__ float h[NB*10];
    __shared__ float A4[10], B4[10];
    int n = threadIdx.x;

    float s[C3];
    #pragma unroll
    for (int c = 0; c < C3; c++) s[c] = (pooled[n*C3+c] >= 0.f) ? 1.f : -1.f;
    #pragma unroll
    for (int f = 0; f < 10; f++){
        float acc = bfc[f];
        #pragma unroll
        for (int c = 0; c < C3; c++) acc += s[c]*fsignf(wfc[f*C3+c]);
        h[n*10+f] = acc;
    }
    __syncthreads();
    if (n < 10){
        float sum = 0.f, ssq = 0.f;
        for (int i = 0; i < NB; i++){ float v = h[i*10+n]; sum += v; ssq += v*v; }
        float m   = sum*(1.0f/NB);
        float var = ssq*(1.0f/NB) - m*m;
        float rs  = rsqrtf(var + 1e-5f);
        float A   = rs*g4[n];
        A4[n] = A; B4[n] = b4[n] - m*A;
    }
    __syncthreads();
    #pragma unroll
    for (int f = 0; f < 10; f++) out[n*10+f] = h[n*10+f]*A4[f] + B4[f];
}

extern "C" void kernel_launch(void* const* d_in, const int* in_sizes, int n_in,
                              void* d_out, int out_size, void* d_ws, size_t ws_size,
                              hipStream_t stream)
{
    const float* x   = (const float*)d_in[0];
    const float* w1  = (const float*)d_in[1];
    const float* w2  = (const float*)d_in[2];
    const float* w3  = (const float*)d_in[3];
    const float* wfc = (const float*)d_in[4];
    const float* bfc = (const float*)d_in[5];
    const float* g1  = (const float*)d_in[6];
    const float* b1  = (const float*)d_in[7];
    const float* g2  = (const float*)d_in[8];
    const float* b2  = (const float*)d_in[9];
    const float* g3  = (const float*)d_in[10];
    const float* b3  = (const float*)d_in[11];
    const float* g4  = (const float*)d_in[12];
    const float* b4  = (const float*)d_in[13];
    float* out = (float*)d_out;
    char*  ws  = (char*)d_ws;

    size_t o_a1    = 0;
    size_t o_a2    = o_a1  + (size_t)NB*OH1*OW1*C1;
    size_t o_bf2   = o_a2  + (size_t)NB*M2*C2;
    size_t o_bf3   = o_bf2 + 4*3*32*32;
    size_t o_stat1 = o_bf3 + 2*9*32*32;
    size_t o_stat2 = o_stat1 + 2*C1*sizeof(float);
    size_t o_stat3 = o_stat2 + 2*C2*sizeof(float);
    size_t o_pool  = o_stat3 + 2*C3*sizeof(float);
    size_t o_coef1 = o_pool  + (size_t)NB*C3*sizeof(float);
    size_t o_coef2 = o_coef1 + 2*C1*sizeof(float);
    size_t o_coef3 = o_coef2 + 2*C2*sizeof(float);
    size_t zbytes  = o_coef1 - o_stat1;

    int8_t* a1    = (int8_t*)(ws + o_a1);
    int8_t* a2    = (int8_t*)(ws + o_a2);
    int8_t* bf2   = (int8_t*)(ws + o_bf2);
    int8_t* bf3   = (int8_t*)(ws + o_bf3);
    float*  stat1 = (float*)(ws + o_stat1);
    float*  stat2 = (float*)(ws + o_stat2);
    float*  stat3 = (float*)(ws + o_stat3);
    float*  pool  = (float*)(ws + o_pool);
    float*  coef1 = (float*)(ws + o_coef1);
    float*  coef2 = (float*)(ws + o_coef2);
    float*  coef3 = (float*)(ws + o_coef3);

    hipMemsetAsync(ws + o_stat1, 0, zbytes, stream);
    prep_bfrag<<<4, 256, 0, stream>>>(w2, w3, bf2, bf3);

    int g1blocks = (NB*OH1*OW1)/256;
    conv1_kernel<<<g1blocks, 256, 0, stream>>>(x, w1, coef1, stat1, a1, 0);
    finalize_kernel<<<1, 64, 0, stream>>>(stat1, g1, b1, coef1,
                                          1.0f/((float)NB*OH1*OW1), C1);
    conv1_kernel<<<g1blocks, 256, 0, stream>>>(x, w1, coef1, stat1, a1, 1);

    dim3 g2dim((MT2+1)/2, NB);
    conv2_kernel<<<g2dim, 32, 0, stream>>>(a1, bf2, coef2, stat2, a2, 0);
    finalize_kernel<<<1, 64, 0, stream>>>(stat2, g2, b2, coef2,
                                          1.0f/((float)NB*M2), C2);
    conv2_kernel<<<g2dim, 32, 0, stream>>>(a1, bf2, coef2, stat2, a2, 1);

    conv3_kernel<<<dim3(MT3, NB), 32, 0, stream>>>(a2, bf3, coef3, stat3, pool, 0);
    finalize_kernel<<<1, 64, 0, stream>>>(stat3, g3, b3, coef3,
                                          1.0f/((float)NB*M3), C3);
    conv3_kernel<<<dim3(MT3, NB), 32, 0, stream>>>(a2, bf3, coef3, stat3, pool, 1);

    head_kernel<<<1, NB, 0, stream>>>(pool, wfc, bfc, g4, b4, out);
}
